// GateRGCN_21449066676409
// MI455X (gfx1250) — matrix operand order repaired and checked by reading, not compiled, not run on hardware
//
#include <hip/hip_runtime.h>

typedef __attribute__((ext_vector_type(16))) __bf16 v16bf;
typedef __attribute__((ext_vector_type(8)))  float  v8f;

#define TILE_M 128
#define TILE_N 128
#define TILE_K 32
#define LDSP    38   // LDS pitch in bf16 elements (19 dwords, odd -> conflict-free frag reads)

// ---------- helpers ----------
__device__ __forceinline__ unsigned short f2bf(float f) {
    unsigned int u = __float_as_uint(f);
    u += 0x7FFFu + ((u >> 16) & 1u);        // round-to-nearest-even
    return (unsigned short)(u >> 16);
}

__global__ void cvt_f32_to_bf16(const float* __restrict__ s,
                                unsigned short* __restrict__ d, size_t n) {
    size_t i  = (size_t)blockIdx.x * blockDim.x + threadIdx.x;
    size_t st = (size_t)gridDim.x * blockDim.x;
    for (; i < n; i += st) d[i] = f2bf(s[i]);
}

__global__ void zero_f32(float* __restrict__ p, size_t n) {
    size_t i  = (size_t)blockIdx.x * blockDim.x + threadIdx.x;
    size_t st = (size_t)gridDim.x * blockDim.x;
    for (; i < n; i += st) p[i] = 0.0f;
}

// out[row][d] = b[d], one block per row stripe
__global__ void init_rows(float* __restrict__ out, const float* __restrict__ b,
                          int N, int Dd) {
    for (int row = blockIdx.x; row < N; row += gridDim.x) {
        float* o = out + (size_t)row * Dd;
        for (int i = threadIdx.x; i < Dd; i += blockDim.x) o[i] = b[i];
    }
}

// ---------- per-relation segment-sum scatter ----------
__global__ void scatter_rel(const float* __restrict__ x, const int* __restrict__ ei,
                            const int* __restrict__ et, float* __restrict__ sum,
                            float* __restrict__ cnt, int r, int E, int Dd) {
    int e = blockIdx.x;
    if (e >= E) return;
    if (et[e] != r) return;
    int src = ei[e];
    int dst = ei[E + e];
    const float* xs = x + (size_t)src * Dd;
    float* sd       = sum + (size_t)dst * Dd;
    for (int i = threadIdx.x; i < Dd; i += blockDim.x)
        atomicAdd(&sd[i], xs[i]);
    if (threadIdx.x == 0) atomicAdd(&cnt[dst], 1.0f);
}

// mean + bf16 quantize, one row per block stripe (cnt read once per row)
__global__ void mean_to_bf(const float* __restrict__ sum, const float* __restrict__ cnt,
                           unsigned short* __restrict__ out, int N, int Dd) {
    for (int row = blockIdx.x; row < N; row += gridDim.x) {
        float rinv = 1.0f / fmaxf(cnt[row], 1.0f);
        const float* s = sum + (size_t)row * Dd;
        unsigned short* o = out + (size_t)row * Dd;
        for (int i = threadIdx.x; i < Dd; i += blockDim.x) o[i] = f2bf(s[i] * rinv);
    }
}

// ---------- bf16 WMMA GEMM:  C(MxN,f32) += A(MxK,bf16,row) @ B(KxN,bf16,row) ----------
// Requirements used here: K % 32 == 0, N % 128 == 0 (true: K = N = 1024).
// M may be ragged; OOB A rows are CLAMPED (their products only reach unwritten outputs).
__global__ __launch_bounds__(256) void gemm_bf16_acc(
    const unsigned short* __restrict__ A, const unsigned short* __restrict__ B,
    float* __restrict__ C, int M, int N, int K)
{
    __shared__ unsigned short As[2][TILE_M * LDSP];   // As[buf][m][k]
    __shared__ unsigned short Bs[2][TILE_N * LDSP];   // transposed: Bs[buf][n][k]

    const int tid  = threadIdx.x;
    const int wave = tid >> 5;
    const int lane = tid & 31;
    const int bm = blockIdx.x * TILE_M;
    const int bn = blockIdx.y * TILE_N;
    const int wm = (wave & 3) * 32;      // wave's M offset in block tile
    const int wn = (wave >> 2) * 64;     // wave's N offset in block tile
    const int fm = lane & 15;            // fragment row within 16
    const int kh = lane >> 4;            // lane half (K split)

    // --- fixed per-thread staging assignments (2 chunks each of A and B) ---
    // A: chunk c -> row = c>>2, kc = (c&3)*8   (512 chunks)
    const int arow0 = tid >> 2;                const int akc0 = (tid & 3) * 8;
    const int arow1 = (tid + 256) >> 2;        const int akc1 = (tid & 3) * 8;
    const int gra0 = min(bm + arow0, M - 1);   // clamp: branchless, harmless
    const int gra1 = min(bm + arow1, M - 1);
    // B: chunk c -> kr = c>>4, nc = (c&15)*8  (512 chunks)
    const int bkr0 = tid >> 4;                 const int bnc0 = (tid & 15) * 8;
    const int bkr1 = (tid + 256) >> 4;         const int bnc1 = (tid & 15) * 8;

    v8f acc[2][4];
#pragma unroll
    for (int i = 0; i < 2; ++i)
#pragma unroll
        for (int j = 0; j < 4; ++j)
#pragma unroll
            for (int t = 0; t < 8; ++t) acc[i][j][t] = 0.0f;

    uint4 ra0, ra1, rb0, rb1;
    auto load_regs = [&](int k0) {
        ra0 = *reinterpret_cast<const uint4*>(A + (size_t)gra0 * K + k0 + akc0);
        ra1 = *reinterpret_cast<const uint4*>(A + (size_t)gra1 * K + k0 + akc1);
        rb0 = *reinterpret_cast<const uint4*>(B + (size_t)(k0 + bkr0) * N + bn + bnc0);
        rb1 = *reinterpret_cast<const uint4*>(B + (size_t)(k0 + bkr1) * N + bn + bnc1);
    };
    auto store_lds = [&](int buf) {
        unsigned int* dp0 = reinterpret_cast<unsigned int*>(&As[buf][arow0 * LDSP + akc0]);
        dp0[0] = ra0.x; dp0[1] = ra0.y; dp0[2] = ra0.z; dp0[3] = ra0.w;
        unsigned int* dp1 = reinterpret_cast<unsigned int*>(&As[buf][arow1 * LDSP + akc1]);
        dp1[0] = ra1.x; dp1[1] = ra1.y; dp1[2] = ra1.z; dp1[3] = ra1.w;
        unsigned short* bp = &Bs[buf][bnc0 * LDSP + bkr0];
        bp[0 * LDSP] = (unsigned short)rb0.x; bp[1 * LDSP] = (unsigned short)(rb0.x >> 16);
        bp[2 * LDSP] = (unsigned short)rb0.y; bp[3 * LDSP] = (unsigned short)(rb0.y >> 16);
        bp[4 * LDSP] = (unsigned short)rb0.z; bp[5 * LDSP] = (unsigned short)(rb0.z >> 16);
        bp[6 * LDSP] = (unsigned short)rb0.w; bp[7 * LDSP] = (unsigned short)(rb0.w >> 16);
        unsigned short* bq = &Bs[buf][bnc1 * LDSP + bkr1];
        bq[0 * LDSP] = (unsigned short)rb1.x; bq[1 * LDSP] = (unsigned short)(rb1.x >> 16);
        bq[2 * LDSP] = (unsigned short)rb1.y; bq[3 * LDSP] = (unsigned short)(rb1.y >> 16);
        bq[4 * LDSP] = (unsigned short)rb1.z; bq[5 * LDSP] = (unsigned short)(rb1.z >> 16);
        bq[6 * LDSP] = (unsigned short)rb1.w; bq[7 * LDSP] = (unsigned short)(rb1.w >> 16);
    };

    // prologue: tile 0 into buffer 0
    load_regs(0);
    store_lds(0);
    __syncthreads();

    const int nk = K / TILE_K;
    for (int it = 0; it < nk; ++it) {
        const int cur = it & 1;
        const bool has_next = (it + 1) < nk;
        if (has_next) load_regs((it + 1) * TILE_K);   // prefetch next tile (global)

        union Frag { unsigned int u[8]; v16bf v; };
        Frag fa[2], fb[4];
        // A fragment: VGPR v -> dword (v&3) + (v>=4)*8 + kh*4   (ISA 16-bit A 16x32 layout)
#pragma unroll
        for (int i = 0; i < 2; ++i) {
            const unsigned int* p =
                reinterpret_cast<const unsigned int*>(&As[cur][(wm + i * 16 + fm) * LDSP]);
            int base = kh * 4;
#pragma unroll
            for (int v = 0; v < 4; ++v) {
                fa[i].u[v]     = p[base + v];
                fa[i].u[v + 4] = p[base + 8 + v];
            }
        }
        // B fragment: VGPR v -> dword (v&3) + (v>=4)*8 + kh*4 (same K order as A)
#pragma unroll
        for (int j = 0; j < 4; ++j) {
            const unsigned int* p =
                reinterpret_cast<const unsigned int*>(&Bs[cur][(wn + j * 16 + fm) * LDSP]);
            int base = kh * 4;
#pragma unroll
            for (int v = 0; v < 8; ++v) fb[j].u[v] = p[base + (v & 3) + (v >> 2) * 8];
        }
#pragma unroll
        for (int i = 0; i < 2; ++i)
#pragma unroll
            for (int j = 0; j < 4; ++j)
                acc[i][j] = __builtin_amdgcn_wmma_f32_16x16x32_bf16(
                    false, fa[i].v, false, fb[j].v, (short)0, acc[i][j], false, false);

        if (has_next) store_lds(1 - cur);             // write other buffer (no conflict)
        __syncthreads();
    }

    // epilogue: C/D layout -> lane: N = lane&15, M = (lane>>4)*8 + vgpr
    const int no = lane & 15;
    const int ms = (lane >> 4) * 8;
#pragma unroll
    for (int i = 0; i < 2; ++i)
#pragma unroll
        for (int j = 0; j < 4; ++j) {
            int gn  = bn + wn + j * 16 + no;          // always < N (N % 128 == 0)
            int gm0 = bm + wm + i * 16 + ms;
#pragma unroll
            for (int t = 0; t < 8; ++t) {
                int gm = gm0 + t;
                if (gm < M) {
                    size_t idx = (size_t)gm * N + gn;
                    C[idx] += acc[i][j][t];
                }
            }
        }
}

// ---------- fused gate: h = tanh(u)*z + x*(1-z), in place over u ----------
__global__ void gate_fuse(float* __restrict__ u_h, const float* __restrict__ x,
                          const float* __restrict__ z, size_t n) {
    size_t i  = (size_t)blockIdx.x * blockDim.x + threadIdx.x;
    size_t st = (size_t)gridDim.x * blockDim.x;
    for (; i < n; i += st) {
        float u  = u_h[i];
        float zz = z[i];
        u_h[i] = tanhf(u) * zz + x[i] * (1.0f - zz);
    }
}

// ---------- launch ----------
extern "C" void kernel_launch(void* const* d_in, const int* in_sizes, int n_in,
                              void* d_out, int out_size, void* d_ws, size_t ws_size,
                              hipStream_t stream) {
    const float* x      = (const float*)d_in[0];
    const int*   eidx   = (const int*)d_in[1];   // [2,E]: src row then dst row
    const int*   etype  = (const int*)d_in[2];
    const float* weight = (const float*)d_in[3]; // [R,D,D]
    const float* root   = (const float*)d_in[4]; // [D,D]
    const float* bias   = (const float*)d_in[5];
    const float* wgate  = (const float*)d_in[6]; // [2D,D]
    const float* bgate  = (const float*)d_in[7];

    const int D = 1024;
    const int N = in_sizes[0] / D;
    const int E = in_sizes[2];
    const int R = in_sizes[3] / (D * D);
    const size_t ND = (size_t)N * D;

    char* ws = (char*)d_ws;
    size_t off = 0;
    auto take = [&](size_t bytes) -> char* {
        char* p = ws + off;
        off = (off + bytes + 255) & ~(size_t)255;
        return p;
    };
    float* sum           = (float*)take(ND * 4 + (size_t)N * 4); // counts appended
    float* cnt           = sum + ND;
    unsigned short* mbf  = (unsigned short*)take(ND * 2);        // mean bf16 / reused as u bf16
    unsigned short* xbf  = (unsigned short*)take(ND * 2);
    unsigned short* wbf  = (unsigned short*)take((size_t)R * D * D * 2);
    unsigned short* rbf  = (unsigned short*)take((size_t)D * D * 2);
    unsigned short* gbf  = (unsigned short*)take((size_t)2 * D * D * 2);
    float* zbuf          = (float*)take(ND * 4);
    (void)ws_size; (void)n_in; (void)out_size;

    float* u = (float*)d_out;

    dim3 blk(256);
    int  cgrid = 2048;
    dim3 ggrid((N + TILE_M - 1) / TILE_M, D / TILE_N);

    // bf16 conversions of dense operands
    cvt_f32_to_bf16<<<cgrid, blk, 0, stream>>>(x, xbf, ND);
    cvt_f32_to_bf16<<<cgrid, blk, 0, stream>>>(weight, wbf, (size_t)R * D * D);
    cvt_f32_to_bf16<<<cgrid, blk, 0, stream>>>(root, rbf, (size_t)D * D);
    cvt_f32_to_bf16<<<cgrid, blk, 0, stream>>>(wgate, gbf, (size_t)2 * D * D);

    // u = bias (rows) ; u += x @ root
    init_rows<<<cgrid, blk, 0, stream>>>(u, bias, N, D);
    gemm_bf16_acc<<<ggrid, blk, 0, stream>>>(xbf, rbf, u, N, D, D);

    // per relation: segment-mean then u += mean @ W_r
    for (int r = 0; r < R; ++r) {
        zero_f32<<<cgrid, blk, 0, stream>>>(sum, ND + (size_t)N);
        scatter_rel<<<E, 256, 0, stream>>>(x, eidx, etype, sum, cnt, r, E, D);
        mean_to_bf<<<cgrid, blk, 0, stream>>>(sum, cnt, mbf, N, D);
        gemm_bf16_acc<<<ggrid, blk, 0, stream>>>(mbf, wbf + (size_t)r * D * D, u, N, D, D);
    }

    // z = b_gate + u @ Wg_top + x @ Wg_bot
    unsigned short* ubf = mbf; // mean buffer free now
    cvt_f32_to_bf16<<<cgrid, blk, 0, stream>>>(u, ubf, ND);
    init_rows<<<cgrid, blk, 0, stream>>>(zbuf, bgate, N, D);
    gemm_bf16_acc<<<ggrid, blk, 0, stream>>>(ubf, gbf, zbuf, N, D, D);
    gemm_bf16_acc<<<ggrid, blk, 0, stream>>>(xbf, gbf + (size_t)D * D, zbuf, N, D, D);

    // h = tanh(u)*z + x*(1-z)  (in place on d_out)
    gate_fuse<<<cgrid, blk, 0, stream>>>(u, x, zbuf, ND);
}
